// SpatialDGCNN_20100446945958
// MI455X (gfx1250) — matrix operand order repaired and checked by reading, not compiled, not run on hardware
//
#include <hip/hip_runtime.h>
#include <hip/hip_bf16.h>
#include <math.h>

// ---------------------------------------------------------------------------
// SpatialDGCNN forward for MI455X (gfx1250, wave32, WMMA 16x16x32 f16).
// All GEMMs (Gram for KNN, graph convs, attention Wh, head conv) run on the
// WMMA pipe with f16 operands / f32 accumulation.
// v3: attention widened to 32 d-columns per iteration (4 accumulators,
// all-lane softmax epilogue); conv/Gram keep 4-accumulator strips.
// ---------------------------------------------------------------------------

#define NPTS   2048
#define BATCH  2
#define KNN    20
#define CATC   1024
#define NHEAD  2
#define NEGBIG (-3.4e38f)

typedef __attribute__((ext_vector_type(16))) _Float16 v16h;
typedef __attribute__((ext_vector_type(8)))  float    v8f;

union HVec { v16h v; _Float16 h[16]; };
union FAcc { v8f  v; float    f[8];  };

__device__ __forceinline__ float lrelu(float x) { return x > 0.f ? x : 0.2f * x; }
__device__ __forceinline__ float elu1 (float x) { return x > 0.f ? x : (__expf(x) - 1.f); }

// BN eval-mode scale: 1/sqrt(1 + 1e-5)
#define BN_SCALE 0.9999950000374997f

// ---------------------------------------------------------------------------
// Layer-1 pairwise "distance" (negative squared distance) on 2D coords.
// pd[b][n][m], diag forced to -inf so top-K excludes self.
// ---------------------------------------------------------------------------
__global__ void pd_l1_kernel(const float* __restrict__ x, float* __restrict__ pd) {
    int n = blockIdx.x, b = blockIdx.y;
    const float* xb = x + (size_t)b * 3 * NPTS;
    float xn = xb[n], yn = xb[NPTS + n];
    float* row = pd + ((size_t)b * NPTS + n) * NPTS;
    for (int m = threadIdx.x; m < NPTS; m += blockDim.x) {
        float dx = xb[m] - xn, dy = xb[NPTS + m] - yn;
        float v = -(dx * dx + dy * dy);
        row[m] = (m == n) ? NEGBIG : v;
    }
}

// ---------------------------------------------------------------------------
// xx[b][n] = sum_c x[c][n]^2 over a channel slice of the concat buffer.
// ---------------------------------------------------------------------------
__global__ void xx_kernel(const float* __restrict__ xcat, int choff, int D,
                          float* __restrict__ xx) {
    int n = blockIdx.x * blockDim.x + threadIdx.x;
    int b = blockIdx.y;
    if (n >= NPTS) return;
    const float* base = xcat + ((size_t)b * CATC + choff) * NPTS + n;
    float s = 0.f;
    for (int c = 0; c < D; c++) {
        float v = base[(size_t)c * NPTS];
        s += v * v;
    }
    xx[b * NPTS + n] = s;
}

// ---------------------------------------------------------------------------
// pd = 2*X^T X - xx[n] - xx[m] via WMMA.  One wave computes a 16(n) x 64(m)
// strip: the n-side A tile is loaded/converted once per k-step and reused
// against 4 m-side B tiles (4 accumulators).
// ---------------------------------------------------------------------------
__global__ void __launch_bounds__(32)
gram_pd_kernel(const float* __restrict__ xcat, int choff, int D,
               const float* __restrict__ xx, float* __restrict__ pd) {
    int lane = threadIdx.x;
    int m0 = blockIdx.x * 64, n0 = blockIdx.y * 16, b = blockIdx.z;
    const float* X = xcat + ((size_t)b * CATC + choff) * NPTS;
    int lrow  = lane & 15;
    int abase = (lane >> 4) * 8;   // A: 16-bit 16x32 layout
    int bbase = (lane >> 4) * 16;  // B: 16-bit 32x16 layout
    v8f acc[4] = {{}, {}, {}, {}};
    for (int c0 = 0; c0 < D; c0 += 32) {
        HVec A;
#pragma unroll
        for (int j = 0; j < 8; j++) {
            A.h[j]     = (_Float16)X[(size_t)(c0 + 2 * abase + j) * NPTS + (n0 + lrow)];
            A.h[j + 8] = (_Float16)X[(size_t)(c0 + 2 * abase + 8 + j) * NPTS + (n0 + lrow)];
        }
#pragma unroll
        for (int t = 0; t < 4; t++) {
            HVec Bv;
#pragma unroll
            for (int j = 0; j < 16; j++)
                Bv.h[j] = (_Float16)X[(size_t)(c0 + bbase + j) * NPTS + (m0 + t * 16 + lrow)];
            acc[t] = __builtin_amdgcn_wmma_f32_16x16x32_f16(false, A.v, false, Bv.v,
                                                            (short)0, acc[t], false, false);
        }
    }
    int rbase = (lane >> 4) * 8;
#pragma unroll
    for (int t = 0; t < 4; t++) {
        FAcc r; r.v = acc[t];
        int col = m0 + t * 16 + (lane & 15);
#pragma unroll
        for (int r8 = 0; r8 < 8; r8++) {
            int n = n0 + rbase + r8;
            float v = 2.f * r.f[r8] - xx[b * NPTS + n] - xx[b * NPTS + col];
            pd[((size_t)b * NPTS + n) * NPTS + col] = (n == col) ? NEGBIG : v;
        }
    }
}

// ---------------------------------------------------------------------------
// Top-K (K=20) neighbor select per row; one wave per (b,n).
// Per-lane sorted insertion lists over strided candidates, then cooperative
// extract-max merge in LDS with __shfl_xor reductions.
// ---------------------------------------------------------------------------
__global__ void __launch_bounds__(32)
knn_select_kernel(const float* __restrict__ pd, int* __restrict__ idxb) {
    __shared__ float sv[32 * KNN];
    __shared__ int   si[32 * KNN];
    int lane = threadIdx.x;
    int n = blockIdx.x, b = blockIdx.y;
    const float* row = pd + ((size_t)b * NPTS + n) * NPTS;
    float lv[KNN]; int li[KNN];
#pragma unroll
    for (int t = 0; t < KNN; t++) { lv[t] = NEGBIG; li[t] = 0x7fffffff; }
    for (int m = lane; m < NPTS; m += 32) {
        __builtin_prefetch(row + m + 512, 0, 1);   // global_prefetch_b8
        float v = row[m];
        if (v > lv[KNN - 1]) {
            int p = KNN - 1;
            while (p > 0 && lv[p - 1] < v) { lv[p] = lv[p - 1]; li[p] = li[p - 1]; p--; }
            lv[p] = v; li[p] = m;
        }
    }
#pragma unroll
    for (int t = 0; t < KNN; t++) { sv[lane * KNN + t] = lv[t]; si[lane * KNN + t] = li[t]; }
    __syncthreads();
    int* orow = idxb + ((size_t)b * NPTS + n) * KNN;
    for (int r = 0; r < KNN; r++) {
        float best = NEGBIG; int bid = 0x7fffffff; int bpos = -1;
        for (int j = lane; j < 32 * KNN; j += 32) {
            float v = sv[j]; int id = si[j];
            if (v > best || (v == best && id < bid)) { best = v; bid = id; bpos = j; }
        }
#pragma unroll
        for (int off = 16; off > 0; off >>= 1) {
            float ov = __shfl_xor(best, off, 32);
            int   oi = __shfl_xor(bid,  off, 32);
            int   op = __shfl_xor(bpos, off, 32);
            if (ov > best || (ov == best && oi < bid)) { best = ov; bid = oi; bpos = op; }
        }
        if (lane == 0) { orow[r] = bid; if (bpos >= 0) sv[bpos] = NEGBIG; }
        __syncthreads();
    }
}

// ---------------------------------------------------------------------------
// stack output: elementwise sorted (min,max) of (neighbor type, center type).
// ---------------------------------------------------------------------------
__global__ void stack_kernel(const float* __restrict__ x, const int* __restrict__ idxb,
                             float* __restrict__ outS) {
    int t = blockIdx.x * blockDim.x + threadIdx.x;
    const int total = BATCH * NPTS * KNN;
    if (t >= total) return;
    int k = t % KNN;
    int n = (t / KNN) % NPTS;
    int b = t / (KNN * NPTS);
    const float* tp = x + (size_t)b * 3 * NPTS + 2 * NPTS;
    int core = (int)tp[n];
    int tg   = (int)tp[idxb[t]];
    int mi = min(core, tg), ma = max(core, tg);
    size_t o = ((size_t)b * NPTS + n) * KNN + k;
    outS[o] = (float)mi;
    outS[(size_t)BATCH * NPTS * KNN + o] = (float)ma;
}

// ---------------------------------------------------------------------------
// Layer-1 features: multi-scale sinusoidal hex embedding diff + center coords.
// feats (B, 32, N, K)
// ---------------------------------------------------------------------------
__global__ void pe_feats_kernel(const float* __restrict__ x, const int* __restrict__ idxb,
                                float* __restrict__ feats) {
    int t = blockIdx.x * blockDim.x + threadIdx.x;
    const int total = BATCH * NPTS * KNN;
    if (t >= total) return;
    int k = t % KNN;
    int n = (t / KNN) % NPTS;
    int b = t / (KNN * NPTS);
    const float* xb = x + (size_t)b * 3 * NPTS;
    int id = idxb[t];
    float nx = xb[id], ny = xb[NPTS + id];
    float cx = xb[n],  cy = xb[NPTS + n];
    const float Ax[3] = {1.f, -0.5f, -0.5f};
    const float Ay[3] = {0.f, -0.8660254037844386f, 0.8660254037844386f};
    const float invS[5] = {1.f, 0.31622776601683794f, 0.1f, 0.031622776601683791f, 0.01f};
    size_t cs   = (size_t)NPTS * KNN;
    size_t base = (size_t)b * 32 * cs + (size_t)n * KNN + k;
#pragma unroll
    for (int d = 0; d < 3; d++) {
        float pn = nx * Ax[d] + ny * Ay[d];
        float pc = cx * Ax[d] + cy * Ay[d];
#pragma unroll
        for (int s = 0; s < 5; s++) {
            float an = pn * invS[s], ac = pc * invS[s];
            feats[base + (size_t)(d * 10 + s) * cs]     = cosf(an) - cosf(ac);
            feats[base + (size_t)(d * 10 + 5 + s) * cs] = sinf(an) - sinf(ac);
        }
    }
    feats[base + 30 * cs] = cx;
    feats[base + 31 * cs] = cy;
}

// ---------------------------------------------------------------------------
// Fused conv_bn_lrelu GEMM: out[b][o][col] = lrelu(BN(sum_c W[o][c]*f[c][col]))
// mode 0: f read from a dense (B, Cin, NK) buffer.
// mode 1: f = graph features gathered on the fly from xcat channel slice:
//         c <  Cp : xp[c][idx[n][k]] - xp[c][n]
//         c >= Cp : xp[c-Cp][n]
// One wave = 64(o) x 16(col) strip: the (expensive, gathered) B tile is
// loaded once per k-step and reused against 4 weight A-tiles.
// ---------------------------------------------------------------------------
__global__ void __launch_bounds__(32)
conv_wmma_kernel(const float* __restrict__ f,
                 const float* __restrict__ xcat, int choff, int Cp,
                 const int* __restrict__ idxb,
                 const float* __restrict__ W, const float* __restrict__ gg,
                 const float* __restrict__ bb,
                 float* __restrict__ out,
                 int Cin, int O, int NK, int mode) {
    int lane = threadIdx.x;
    int col0 = blockIdx.x * 16, o0 = blockIdx.y * 64, b = blockIdx.z;
    int lcol  = col0 + (lane & 15);
    int lrow  = lane & 15;
    int abase = (lane >> 4) * 8;
    int bbase = (lane >> 4) * 16;

    int n = 0, id = 0;
    const float* xb = xcat + ((size_t)b * CATC + choff) * NPTS;
    if (mode == 1) {
        n = lcol / KNN;
        int kk = lcol - n * KNN;
        id = idxb[((size_t)b * NPTS + n) * KNN + kk];
    }
    const float* fb = f + (size_t)b * Cin * NK;

    v8f acc[4] = {{}, {}, {}, {}};
    for (int c0 = 0; c0 < Cin; c0 += 32) {
        HVec Bv;
        if (mode == 0) {
#pragma unroll
            for (int j = 0; j < 16; j++)
                Bv.h[j] = (_Float16)fb[(size_t)(c0 + bbase + j) * NK + lcol];
        } else {
#pragma unroll
            for (int j = 0; j < 16; j++) {
                int c = c0 + bbase + j;
                float v;
                if (c < Cp) v = xb[(size_t)c * NPTS + id] - xb[(size_t)c * NPTS + n];
                else        v = xb[(size_t)(c - Cp) * NPTS + n];
                Bv.h[j] = (_Float16)v;
            }
        }
#pragma unroll
        for (int t = 0; t < 4; t++) {
            const float* wr = W + (size_t)(o0 + t * 16 + lrow) * Cin + c0;
            HVec A;
#pragma unroll
            for (int j = 0; j < 8; j++) {
                A.h[j]     = (_Float16)wr[2 * abase + j];
                A.h[j + 8] = (_Float16)wr[2 * abase + 8 + j];
            }
            acc[t] = __builtin_amdgcn_wmma_f32_16x16x32_f16(false, A.v, false, Bv.v,
                                                            (short)0, acc[t], false, false);
        }
    }
    int rbase = (lane >> 4) * 8;
    int wcol  = col0 + (lane & 15);
#pragma unroll
    for (int t = 0; t < 4; t++) {
        FAcc r; r.v = acc[t];
#pragma unroll
        for (int r8 = 0; r8 < 8; r8++) {
            int o = o0 + t * 16 + rbase + r8;
            float v = r.f[r8] * BN_SCALE * gg[o] + bb[o];
            out[((size_t)b * O + o) * NK + wcol] = lrelu(v);
        }
    }
}

// ---------------------------------------------------------------------------
// Fused typed-edge attention head:
//   Wh = x(20 x C) * Wm(C x C)  via WMMA (A in LDS, rows padded to 32),
//   e = lrelu(Wh * a_typed), softmax over the 20 neighbors, out = elu(att*Wh),
//   ax (optional) written to global, mean over k fused into xcat slice.
// One wave per (n, head, b).  32 d-columns per iteration: two B tiles against
// the LDS A row-tiles (4 WMMAs), then an all-lane softmax epilogue (one
// column per lane).
// ---------------------------------------------------------------------------
__global__ void __launch_bounds__(32)
attn_kernel(const float* __restrict__ h, const float* __restrict__ x,
            const int* __restrict__ idxb,
            const float* __restrict__ aw, const float* __restrict__ ap,
            const float* __restrict__ asf,
            float* __restrict__ ax_out,            // may be null
            float* __restrict__ xcat, int choff, int C) {
    __shared__ _Float16 Ah[32][256];
    __shared__ float    wh[32][33];   // 32 k-rows x 32 d-cols (+1 pad)
    __shared__ int      prow[KNN];
    int lane = threadIdx.x;
    int n = blockIdx.x, head = blockIdx.y, b = blockIdx.z;

    const float* tp = x + (size_t)b * 3 * NPTS + 2 * NPTS;
    int core = (int)tp[n];
    if (lane > 0 && lane < KNN) {
        int id = idxb[((size_t)b * NPTS + n) * KNN + lane];
        int tg = (int)tp[id];
        int mi = min(core, tg), ma = max(core, tg);
        prow[lane] = mi * 8 - (mi * (mi - 1)) / 2 + (ma - mi);   // PAIR_LUT row
    }
    const float* hb = h + (size_t)b * C * NPTS * KNN + (size_t)n * KNN;
    for (int k = 0; k < 32; k++)
        for (int c = lane; c < C; c += 32)
            Ah[k][c] = (k < KNN) ? (_Float16)hb[(size_t)c * NPTS * KNN + k] : (_Float16)0.f;
    __syncthreads();

    const float* awh = aw + (size_t)head * C * C;
    int lrow  = lane & 15;
    int abase = (lane >> 4) * 8;
    int bbase = (lane >> 4) * 16;

    for (int d0 = 0; d0 < C; d0 += 32) {
        v8f acc00 = {}, acc10 = {}, acc01 = {}, acc11 = {};
        for (int c0 = 0; c0 < C; c0 += 32) {
            HVec A0, A1, Bv0, Bv1;
#pragma unroll
            for (int j = 0; j < 8; j++) {
                A0.h[j]     = Ah[lrow][c0 + 2 * abase + j];
                A0.h[j + 8] = Ah[lrow][c0 + 2 * abase + 8 + j];
                A1.h[j]     = Ah[16 + lrow][c0 + 2 * abase + j];
                A1.h[j + 8] = Ah[16 + lrow][c0 + 2 * abase + 8 + j];
            }
#pragma unroll
            for (int j = 0; j < 16; j++) {
                const float* wrow = awh + (size_t)(c0 + bbase + j) * C + d0 + lrow;
                Bv0.h[j] = (_Float16)wrow[0];
                Bv1.h[j] = (_Float16)wrow[16];
            }
            acc00 = __builtin_amdgcn_wmma_f32_16x16x32_f16(false, A0.v, false, Bv0.v,
                                                           (short)0, acc00, false, false);
            acc10 = __builtin_amdgcn_wmma_f32_16x16x32_f16(false, A1.v, false, Bv0.v,
                                                           (short)0, acc10, false, false);
            acc01 = __builtin_amdgcn_wmma_f32_16x16x32_f16(false, A0.v, false, Bv1.v,
                                                           (short)0, acc01, false, false);
            acc11 = __builtin_amdgcn_wmma_f32_16x16x32_f16(false, A1.v, false, Bv1.v,
                                                           (short)0, acc11, false, false);
        }
        FAcc r00, r10, r01, r11;
        r00.v = acc00; r10.v = acc10; r01.v = acc01; r11.v = acc11;
        int wc = lane & 15;
        int rb = (lane >> 4) * 8;
#pragma unroll
        for (int r8 = 0; r8 < 8; r8++) {
            wh[rb + r8][wc]           = r00.f[r8];
            wh[16 + rb + r8][wc]      = r10.f[r8];
            wh[rb + r8][16 + wc]      = r01.f[r8];
            wh[16 + rb + r8][16 + wc] = r11.f[r8];
        }
        __syncthreads();
        {
            int d = d0 + lane;                    // 32 columns, one per lane
            float e[KNN];
            float mx = NEGBIG;
#pragma unroll
            for (int k = 0; k < KNN; k++) {
                float a = (k == 0) ? asf[((size_t)head * 8 + core) * C + d]
                                   : ap[((size_t)head * 36 + prow[k]) * C + d];
                float ee = lrelu(wh[k][lane] * a);
                e[k] = ee;
                mx = fmaxf(mx, ee);
            }
            float s = 0.f;
#pragma unroll
            for (int k = 0; k < KNN; k++) { e[k] = __expf(e[k] - mx); s += e[k]; }
            float inv = 1.f / s;
            float msum = 0.f;
#pragma unroll
            for (int k = 0; k < KNN; k++) {
                float o = elu1(e[k] * inv * wh[k][lane]);
                if (ax_out)
                    ax_out[(((size_t)b * (2 * C) + (size_t)head * C + d) * NPTS + n) * KNN + k] = o;
                msum += o;
            }
            xcat[((size_t)b * CATC + choff + head * C + d) * NPTS + n] = msum * (1.0f / KNN);
        }
        __syncthreads();
    }
}

// ---------------------------------------------------------------------------
// Global max+mean pool over points: p[b][c]=max, p[b][1024+c]=mean.
// ---------------------------------------------------------------------------
__global__ void pool_kernel(const float* __restrict__ x5, float* __restrict__ p) {
    int c = blockIdx.x, b = blockIdx.y, tid = threadIdx.x;
    const float* v = x5 + ((size_t)b * CATC + c) * NPTS;
    float mx = NEGBIG, sm = 0.f;
    for (int n = tid; n < NPTS; n += blockDim.x) { float q = v[n]; mx = fmaxf(mx, q); sm += q; }
    __shared__ float smx[256], ssm[256];
    smx[tid] = mx; ssm[tid] = sm; __syncthreads();
    for (int s = 128; s > 0; s >>= 1) {
        if (tid < s) { smx[tid] = fmaxf(smx[tid], smx[tid + s]); ssm[tid] += ssm[tid + s]; }
        __syncthreads();
    }
    if (tid == 0) {
        p[(size_t)b * 2048 + c]        = smx[0];
        p[(size_t)b * 2048 + 1024 + c] = ssm[0] / (float)NPTS;
    }
}

// ---------------------------------------------------------------------------
// Classifier MLP: 2048 -> 512(bn,lrelu) -> 256(bias,bn,lrelu) -> 2.
// ---------------------------------------------------------------------------
__global__ void mlp_kernel(const float* __restrict__ p,
                           const float* __restrict__ l1w, const float* __restrict__ g6,
                           const float* __restrict__ b6,
                           const float* __restrict__ l2w, const float* __restrict__ l2b,
                           const float* __restrict__ g7, const float* __restrict__ b7,
                           const float* __restrict__ l3w, const float* __restrict__ l3b,
                           float* __restrict__ y) {
    __shared__ float pp[2048];
    __shared__ float y1[512];
    __shared__ float y2[256];
    int b = blockIdx.x, tid = threadIdx.x;
    for (int i = tid; i < 2048; i += 256) pp[i] = p[(size_t)b * 2048 + i];
    __syncthreads();
    for (int o = tid; o < 512; o += 256) {
        float s = 0.f;
        for (int i = 0; i < 2048; i++) s += pp[i] * l1w[(size_t)o * 2048 + i];
        y1[o] = lrelu(s * BN_SCALE * g6[o] + b6[o]);
    }
    __syncthreads();
    for (int o = tid; o < 256; o += 256) {
        float s = l2b[o];
        for (int i = 0; i < 512; i++) s += y1[i] * l2w[(size_t)o * 512 + i];
        y2[o] = lrelu(s * BN_SCALE * g7[o] + b7[o]);
    }
    __syncthreads();
    if (tid < 2) {
        float s = l3b[tid];
        for (int i = 0; i < 256; i++) s += y2[i] * l3w[(size_t)tid * 256 + i];
        y[b * 2 + tid] = s;
    }
}

// ---------------------------------------------------------------------------
extern "C" void kernel_launch(void* const* d_in, const int* in_sizes, int n_in,
                              void* d_out, int out_size, void* d_ws, size_t ws_size,
                              hipStream_t stream) {
    (void)in_sizes; (void)n_in; (void)out_size; (void)ws_size;
    const float* x   = (const float*)d_in[0];
    const float* w1  = (const float*)d_in[1];
    const float* g1  = (const float*)d_in[2];
    const float* b1  = (const float*)d_in[3];
    const float* w2  = (const float*)d_in[4];
    const float* g2  = (const float*)d_in[5];
    const float* b2  = (const float*)d_in[6];
    const float* w3  = (const float*)d_in[7];
    const float* g3  = (const float*)d_in[8];
    const float* b3  = (const float*)d_in[9];
    const float* w4  = (const float*)d_in[10];
    const float* g4  = (const float*)d_in[11];
    const float* b4  = (const float*)d_in[12];
    const float* w5  = (const float*)d_in[13];
    const float* g5  = (const float*)d_in[14];
    const float* b5  = (const float*)d_in[15];
    const float* aw1 = (const float*)d_in[16];
    const float* ap1 = (const float*)d_in[17];
    const float* as1 = (const float*)d_in[18];
    const float* aw2 = (const float*)d_in[19];
    const float* ap2 = (const float*)d_in[20];
    const float* as2 = (const float*)d_in[21];
    const float* aw3 = (const float*)d_in[22];
    const float* ap3 = (const float*)d_in[23];
    const float* as3 = (const float*)d_in[24];
    const float* aw4 = (const float*)d_in[25];
    const float* ap4 = (const float*)d_in[26];
    const float* as4 = (const float*)d_in[27];
    const float* l1w = (const float*)d_in[28];
    const float* g6  = (const float*)d_in[29];
    const float* b6  = (const float*)d_in[30];
    const float* l2w = (const float*)d_in[31];
    const float* l2b = (const float*)d_in[32];
    const float* g7  = (const float*)d_in[33];
    const float* b7  = (const float*)d_in[34];
    const float* l3w = (const float*)d_in[35];
    const float* l3b = (const float*)d_in[36];

    // -------- workspace layout (floats) --------
    float* ws = (float*)d_ws;
    const size_t NKf = (size_t)NPTS * KNN;                  // 40960
    float* pd     = ws;                                     // B*N*N (also reused for feats1)
    float* feats1 = ws;                                     // overlays pd (disjoint lifetime)
    float* xxb    = ws + (size_t)BATCH * NPTS * NPTS;       // B*N
    int*   idxb   = (int*)(xxb + (size_t)BATCH * NPTS);     // B*N*K ints
    float* hbuf   = (float*)(idxb + (size_t)BATCH * NKf);   // B*256*N*K max
    float* xcat   = hbuf + (size_t)BATCH * 256 * NKf;       // B*1024*N
    float* x5b    = xcat + (size_t)BATCH * CATC * NPTS;     // B*1024*N
    float* pbuf   = x5b  + (size_t)BATCH * CATC * NPTS;     // B*2048

    float* yout = (float*)d_out;
    float* ax4  = yout + 4;
    float* stk  = ax4 + (size_t)BATCH * 512 * NKf;

    const int NKtot = NPTS * KNN;
    const int total = BATCH * NPTS * KNN;

    // ---------------- Layer 1 (spatial KNN + position embedding) -----------
    pd_l1_kernel<<<dim3(NPTS, BATCH), 256, 0, stream>>>(x, pd);
    knn_select_kernel<<<dim3(NPTS, BATCH), 32, 0, stream>>>(pd, idxb);
    stack_kernel<<<(total + 255) / 256, 256, 0, stream>>>(x, idxb, stk);
    pe_feats_kernel<<<(total + 255) / 256, 256, 0, stream>>>(x, idxb, feats1);
    conv_wmma_kernel<<<dim3(NKtot / 16, 64 / 64, BATCH), 32, 0, stream>>>(
        feats1, xcat, 0, 0, idxb, w1, g1, b1, hbuf, 32, 64, NKtot, 0);
    attn_kernel<<<dim3(NPTS, NHEAD, BATCH), 32, 0, stream>>>(
        hbuf, x, idxb, aw1, ap1, as1, nullptr, xcat, 0, 64);

    // ---------------- Layer 2 (dyn: KNN on x1, 128ch) -----------------------
    xx_kernel<<<dim3(NPTS / 256, BATCH), 256, 0, stream>>>(xcat, 0, 128, xxb);
    gram_pd_kernel<<<dim3(NPTS / 64, NPTS / 16, BATCH), 32, 0, stream>>>(xcat, 0, 128, xxb, pd);
    knn_select_kernel<<<dim3(NPTS, BATCH), 32, 0, stream>>>(pd, idxb);
    conv_wmma_kernel<<<dim3(NKtot / 16, 64 / 64, BATCH), 32, 0, stream>>>(
        xcat, xcat, 0, 128, idxb, w2, g2, b2, hbuf, 256, 64, NKtot, 1);
    attn_kernel<<<dim3(NPTS, NHEAD, BATCH), 32, 0, stream>>>(
        hbuf, x, idxb, aw2, ap2, as2, nullptr, xcat, 128, 64);

    // ---------------- Layer 3 ------------------------------------------------
    xx_kernel<<<dim3(NPTS / 256, BATCH), 256, 0, stream>>>(xcat, 128, 128, xxb);
    gram_pd_kernel<<<dim3(NPTS / 64, NPTS / 16, BATCH), 32, 0, stream>>>(xcat, 128, 128, xxb, pd);
    knn_select_kernel<<<dim3(NPTS, BATCH), 32, 0, stream>>>(pd, idxb);
    conv_wmma_kernel<<<dim3(NKtot / 16, 128 / 64, BATCH), 32, 0, stream>>>(
        xcat, xcat, 128, 128, idxb, w3, g3, b3, hbuf, 256, 128, NKtot, 1);
    attn_kernel<<<dim3(NPTS, NHEAD, BATCH), 32, 0, stream>>>(
        hbuf, x, idxb, aw3, ap3, as3, nullptr, xcat, 256, 128);

    // ---------------- Layer 4 (ax4 straight to d_out) -----------------------
    xx_kernel<<<dim3(NPTS / 256, BATCH), 256, 0, stream>>>(xcat, 256, 256, xxb);
    gram_pd_kernel<<<dim3(NPTS / 64, NPTS / 16, BATCH), 32, 0, stream>>>(xcat, 256, 256, xxb, pd);
    knn_select_kernel<<<dim3(NPTS, BATCH), 32, 0, stream>>>(pd, idxb);
    conv_wmma_kernel<<<dim3(NKtot / 16, 256 / 64, BATCH), 32, 0, stream>>>(
        xcat, xcat, 256, 256, idxb, w4, g4, b4, hbuf, 512, 256, NKtot, 1);
    attn_kernel<<<dim3(NPTS, NHEAD, BATCH), 32, 0, stream>>>(
        hbuf, x, idxb, aw4, ap4, as4, ax4, xcat, 512, 256);

    // ---------------- Head: 1024x1024 conv, pool, MLP -----------------------
    conv_wmma_kernel<<<dim3(NPTS / 16, CATC / 64, BATCH), 32, 0, stream>>>(
        xcat, xcat, 0, 0, idxb, w5, g5, b5, x5b, 1024, 1024, NPTS, 0);
    pool_kernel<<<dim3(CATC, BATCH), 256, 0, stream>>>(x5b, pbuf);
    mlp_kernel<<<BATCH, 256, 0, stream>>>(pbuf, l1w, g6, b6, l2w, l2b, g7, b7, l3w, l3b, yout);
}
